// PointConv_85177791414870
// MI455X (gfx1250) — matrix operand order repaired and checked by reading, not compiled, not run on hardware
//
#include <hip/hip_runtime.h>

typedef __attribute__((ext_vector_type(16))) _Float16 v16h;
typedef __attribute__((ext_vector_type(8)))  float    v8f;

#define BATCH 4
#define NAG   512
#define HDIM  64
#define F2D   128
#define WAVES 8
#define JT    16
#define SROW  68          // padded row stride in u32 (64 data + 4 pad -> conflict-free)

#define USE_ASYNC_LDS 1   // gfx1250 GLOBAL_LOAD_ASYNC_TO_LDS path

union Frag { uint4 q[2]; v16h h; };

static __device__ __forceinline__ unsigned pack_h2(float a, float b) {
  _Float16 ha = (_Float16)a, hb = (_Float16)b;
  unsigned short ua = __builtin_bit_cast(unsigned short, ha);
  unsigned short ub = __builtin_bit_cast(unsigned short, hb);
  return (unsigned)ua | ((unsigned)ub << 16);
}

#if USE_ASYNC_LDS
// Memory -> LDS, 16B per lane, tracked by ASYNCcnt (no VGPR round-trip).
// Generic LDS pointers hold the LDS byte offset in addr[31:0] (ISA 10.2).
static __device__ __forceinline__ void async_cp_b128(unsigned lds_off, const void* gptr) {
  unsigned long long ga = (unsigned long long)(uintptr_t)gptr;
  asm volatile("global_load_async_to_lds_b128 %0, %1, off"
               :: "v"(lds_off), "v"(ga) : "memory");
}
static __device__ __forceinline__ void async_wait0() {
  asm volatile("s_wait_asynccnt 0" ::: "memory");
}
#endif

static __device__ __forceinline__ _Float16 relu16(_Float16 x) {
#if __has_builtin(__builtin_fmaxf16)
  return __builtin_fmaxf16(x, (_Float16)0.0f);
#else
  return x > (_Float16)0 ? x : (_Float16)0;
#endif
}

// ---------------------------------------------------------------------------
// Kernel 1a: build per-agent vectors
//   s[b,j,f] = (x_j . wx)[f] - (pos_j . wd)[f] + b1[f]   (A-fragment f16 layout)
//   p[b,i,f] = (pos_i . wd)[f]                            (same layout)
// A-frag pair order (per u32 index u in 0..63):
//   g=u/8, r=u%8, kk=g/2, half=g%2,
//   f0 = 32*kk + 8*half + (r<4 ? 2r : 16 + 2(r-4)), pair (f0, f0+1)
// ---------------------------------------------------------------------------
__global__ void pack_sp_kernel(const float* __restrict__ xp, const float* __restrict__ xy,
                               const float* __restrict__ xo, const float* __restrict__ pp,
                               const float* __restrict__ py, const float* __restrict__ po,
                               const float* __restrict__ w1, const float* __restrict__ b1,
                               unsigned* __restrict__ s_packed, unsigned* __restrict__ p_packed) {
  int idx = blockIdx.x * blockDim.x + threadIdx.x;     // exactly B*512*64 threads
  int b = idx >> 15;
  int j = (idx >> 6) & 511;
  int u = idx & 63;
  int g = u >> 3, r = u & 7;
  int kk = g >> 1, hf = g & 1;
  int f0 = kk*32 + hf*8 + (r < 4 ? 2*r : 16 + 2*(r-4));

  const float* xrow; const float* prow;
  if (j < 64)       { xrow = xp + ((size_t)b*64  + j)       * HDIM; prow = pp + ((size_t)b*64  + j)       * 2; }
  else if (j < 192) { xrow = xy + ((size_t)b*128 + (j-64))  * HDIM; prow = py + ((size_t)b*128 + (j-64))  * 2; }
  else              { xrow = xo + ((size_t)b*320 + (j-192)) * HDIM; prow = po + ((size_t)b*320 + (j-192)) * 2; }

  float xw0 = 0.f, xw1 = 0.f;
  #pragma unroll 8
  for (int h = 0; h < HDIM; ++h) {
    float xv = xrow[h];
    xw0 = fmaf(xv, w1[h*F2D + f0],     xw0);
    xw1 = fmaf(xv, w1[h*F2D + f0 + 1], xw1);
  }
  float p0 = prow[0], p1 = prow[1];
  float pw0 = fmaf(p0, w1[64*F2D + f0],     p1 * w1[65*F2D + f0]);
  float pw1 = fmaf(p0, w1[64*F2D + f0 + 1], p1 * w1[65*F2D + f0 + 1]);
  float s0 = xw0 - pw0 + b1[f0];
  float s1 = xw1 - pw1 + b1[f0 + 1];
  s_packed[idx] = pack_h2(s0, s1);
  p_packed[idx] = pack_h2(pw0, pw1);
}

// ---------------------------------------------------------------------------
// Kernel 1b: pack w2 (128x64, row-major [f][n]) into B-fragment f16 layout.
// B-frag pair order per column n (u in 0..63):
//   g=u/8, r=u%8, kk=g/2, hB=g%2, f0 = 32*kk + 8*hB + (r<4 ? 2r : 16 + 2(r-4))
// ---------------------------------------------------------------------------
__global__ void pack_w2_kernel(const float* __restrict__ w2, unsigned* __restrict__ w2p) {
  int idx = blockIdx.x * blockDim.x + threadIdx.x;     // 64*64 threads
  int n = idx >> 6, u = idx & 63;
  int g = u >> 3, r = u & 7;
  int kk = g >> 1, hB = g & 1;
  int f0 = kk*32 + hB*8 + (r < 4 ? 2*r : 16 + 2*(r-4));
  w2p[idx] = pack_h2(w2[f0*HDIM + n], w2[(f0+1)*HDIM + n]);
}

// ---------------------------------------------------------------------------
// Main kernel: one wave per agent i, 8 waves per WG sharing s[b] in LDS.
// Per j-tile of 16: A = relu(s_tile + p_i) (f16), C[nn] += A x Bf[kk][nn]
// via v_wmma_f32_16x16x32_f16; reduce C over j into min/max/sum; tail MLP.
// ---------------------------------------------------------------------------
__global__ __launch_bounds__(256) void pointconv_main(
    const unsigned* __restrict__ s_packed, const unsigned* __restrict__ p_packed,
    const unsigned* __restrict__ w2p,
    const float* __restrict__ b2, const float* __restrict__ w3, const float* __restrict__ b3,
    const float* __restrict__ w4, const float* __restrict__ b4,
    const float* __restrict__ xp, const float* __restrict__ xy, const float* __restrict__ xo,
    float* __restrict__ out) {
  extern __shared__ unsigned char smem[];
  unsigned* slds   = (unsigned*)smem;                       // NAG rows * SROW u32
  float*    aggbuf = (float*)(smem + NAG*SROW*4);           // WAVES*192
  float*    hidbuf = aggbuf + WAVES*192;                    // WAVES*128

  const int b    = blockIdx.y;
  const int wave = threadIdx.x >> 5;
  const int lane = threadIdx.x & 31;
  const int i    = blockIdx.x * WAVES + wave;
  const int lh   = lane & 15;
  const int hs   = lane >> 4;

  // stage s[b] (512 x 64 u32) into padded LDS rows
  {
    const unsigned* src = s_packed + (size_t)b * NAG * 64;
    for (int t = threadIdx.x; t < NAG*16; t += 256) {
      int row = t >> 4, c4 = t & 15;
#if USE_ASYNC_LDS
      async_cp_b128((unsigned)(uintptr_t)(slds + row*SROW + c4*4),
                    src + row*64 + c4*4);
#else
      uint4 v = ((const uint4*)src)[row*16 + c4];
      *((uint4*)(slds + row*SROW + c4*4)) = v;
#endif
    }
  }

  // warm L2 for the tail-MLP weights
  for (int t = threadIdx.x; t < 768; t += 256) __builtin_prefetch(w3 + t*32, 0, 1);
  for (int t = threadIdx.x; t < 256; t += 256) __builtin_prefetch(w4 + t*32, 0, 1);

  // p-fragments for this wave's i (resident)
  Frag pf[4];
  {
    const uint4* psrc = (const uint4*)(p_packed + ((size_t)b*NAG + i) * 64);
    #pragma unroll
    for (int kk = 0; kk < 4; ++kk) {
      int g = kk*2 + hs;
      pf[kk].q[0] = psrc[g*2 + 0];
      pf[kk].q[1] = psrc[g*2 + 1];
    }
  }
  // B-fragments (w2) resident: 4 K-steps x 4 N-tiles
  Frag Bf[4][4];
  #pragma unroll
  for (int nn = 0; nn < 4; ++nn) {
    const uint4* bsrc = (const uint4*)(w2p + (nn*16 + lh) * 64);
    #pragma unroll
    for (int kk = 0; kk < 4; ++kk) {
      int g = kk*2 + hs;
      Bf[kk][nn].q[0] = bsrc[g*2 + 0];
      Bf[kk][nn].q[1] = bsrc[g*2 + 1];
    }
  }

#if USE_ASYNC_LDS
  async_wait0();
#endif
  __syncthreads();

  float rmin[4], rmax[4], rsum[4];
  #pragma unroll
  for (int nn = 0; nn < 4; ++nn) { rmin[nn] = 3.0e38f; rmax[nn] = -3.0e38f; rsum[nn] = 0.f; }

  for (int j0 = 0; j0 < NAG; j0 += JT) {
    v8f acc[4] = {};
    #pragma unroll
    for (int kk = 0; kk < 4; ++kk) {
      Frag sf;
      const uint4* sp = (const uint4*)(slds + (j0 + lh)*SROW + (kk*2 + hs)*8);
      sf.q[0] = sp[0];
      sf.q[1] = sp[1];
      v16h hv = sf.h + pf[kk].h;            // packed f16 add
      #pragma unroll
      for (int e = 0; e < 16; ++e)          // packed f16 relu
        hv[e] = relu16(hv[e]);
      #pragma unroll
      for (int nn = 0; nn < 4; ++nn)
        acc[nn] = __builtin_amdgcn_wmma_f32_16x16x32_f16(
            false, hv, false, Bf[kk][nn].h, (short)0, acc[nn], false, false);
    }
    // reduce this 16-j tile into running stats
    #pragma unroll
    for (int nn = 0; nn < 4; ++nn) {
      float mn = acc[nn][0], mx = acc[nn][0], sm = acc[nn][0];
      #pragma unroll
      for (int e = 1; e < 8; ++e) {
        float v = acc[nn][e];
        mn = fminf(mn, v); mx = fmaxf(mx, v); sm += v;
      }
      rmin[nn] = fminf(rmin[nn], mn);
      rmax[nn] = fmaxf(rmax[nn], mx);
      rsum[nn] += sm;
    }
  }

  // combine lane halves (rows 0-7 vs 8-15 of each C tile)
  #pragma unroll
  for (int nn = 0; nn < 4; ++nn) {
    rmin[nn] = fminf(rmin[nn], __shfl_xor(rmin[nn], 16));
    rmax[nn] = fmaxf(rmax[nn], __shfl_xor(rmax[nn], 16));
    rsum[nn] += __shfl_xor(rsum[nn], 16);
  }

  float* aggw = aggbuf + wave * 192;
  if (lane < 16) {
    #pragma unroll
    for (int nn = 0; nn < 4; ++nn) {
      int n = nn*16 + lane;
      float bb = b2[n];
      aggw[n]       = rmin[nn] + bb;
      aggw[64 + n]  = rmax[nn] + bb;
      aggw[128 + n] = rsum[nn] * (1.0f / NAG) + bb;
    }
  }

  __syncthreads();   // everyone done with slds; agg visible

  // reuse s region for w3 (192x128) and w4 (128x64)
  float* w3l = (float*)smem;
  float* w4l = w3l + 192*128;
#if USE_ASYNC_LDS
  for (int t = threadIdx.x; t < (192*128)/4; t += 256)
    async_cp_b128((unsigned)(uintptr_t)(w3l + t*4), w3 + t*4);
  for (int t = threadIdx.x; t < (128*64)/4; t += 256)
    async_cp_b128((unsigned)(uintptr_t)(w4l + t*4), w4 + t*4);
  async_wait0();
#else
  for (int t = threadIdx.x; t < 192*128; t += 256) w3l[t] = w3[t];
  for (int t = threadIdx.x; t < 128*64;  t += 256) w4l[t] = w4[t];
#endif
  __syncthreads();

  // hidden = relu(agg @ w3 + b3), 4 units per lane
  float* hidw = hidbuf + wave * 128;
  #pragma unroll
  for (int r = 0; r < 4; ++r) {
    int f = lane + 32*r;
    float a = b3[f];
    #pragma unroll 4
    for (int k = 0; k < 192; ++k) a = fmaf(aggw[k], w3l[k*128 + f], a);
    hidw[f] = fmaxf(a, 0.f);
  }
  __syncthreads();

  // out = x_i + hidden @ w4 + b4, 2 units per lane
  const float* xrow; size_t obase;
  if (i < 64)       { xrow = xp + ((size_t)b*64  + i)       * HDIM; obase = ((size_t)b*64 + i) * HDIM; }
  else if (i < 192) { xrow = xy + ((size_t)b*128 + (i-64))  * HDIM;
                      obase = (size_t)BATCH*64*HDIM + ((size_t)b*128 + (i-64)) * HDIM; }
  else              { xrow = xo + ((size_t)b*320 + (i-192)) * HDIM;
                      obase = (size_t)BATCH*64*HDIM + (size_t)BATCH*128*HDIM
                            + ((size_t)b*320 + (i-192)) * HDIM; }
  #pragma unroll
  for (int r = 0; r < 2; ++r) {
    int h = lane + 32*r;
    float o = b4[h];
    #pragma unroll 4
    for (int k = 0; k < 128; ++k) o = fmaf(hidw[k], w4l[k*64 + h], o);
    out[obase + h] = xrow[h] + o;
  }
}

// ---------------------------------------------------------------------------
extern "C" void kernel_launch(void* const* d_in, const int* in_sizes, int n_in,
                              void* d_out, int out_size, void* d_ws, size_t ws_size,
                              hipStream_t stream) {
  const float* xp = (const float*)d_in[0];
  const float* xy = (const float*)d_in[1];
  const float* xo = (const float*)d_in[2];
  const float* pp = (const float*)d_in[3];
  const float* py = (const float*)d_in[4];
  const float* po = (const float*)d_in[5];
  const float* w1 = (const float*)d_in[6];
  const float* b1 = (const float*)d_in[7];
  const float* w2 = (const float*)d_in[8];
  const float* b2 = (const float*)d_in[9];
  const float* w3 = (const float*)d_in[10];
  const float* b3 = (const float*)d_in[11];
  const float* w4 = (const float*)d_in[12];
  const float* b4 = (const float*)d_in[13];

  unsigned* s_packed = (unsigned*)d_ws;                      // B*512*64 u32
  unsigned* p_packed = s_packed + (size_t)BATCH*NAG*64;      // B*512*64 u32
  unsigned* w2p      = p_packed + (size_t)BATCH*NAG*64;      // 64*64 u32

  pack_sp_kernel<<<dim3((BATCH*NAG*64)/256), 256, 0, stream>>>(
      xp, xy, xo, pp, py, po, w1, b1, s_packed, p_packed);
  pack_w2_kernel<<<dim3((64*64)/256), 256, 0, stream>>>(w2, w2p);

  size_t smem = (size_t)NAG*SROW*4 + (size_t)(WAVES*192 + WAVES*128)*4;
  pointconv_main<<<dim3(NAG/WAVES, BATCH), 256, smem, stream>>>(
      s_packed, p_packed, w2p, b2, w3, b3, w4, b4, xp, xy, xo, (float*)d_out);
}
